// GraphUnet_84215718740464
// MI455X (gfx1250) — matrix operand order repaired and checked by reading, not compiled, not run on hardware
//
#include <hip/hip_runtime.h>
#include <hip/hip_bf16.h>

typedef __attribute__((ext_vector_type(16))) _Float16 v16h;
typedef __attribute__((ext_vector_type(8)))  _Float16 v8h;
typedef __attribute__((ext_vector_type(8)))  float    v8f;

#define TPB 256
#define MAXH 4

// ---------------------------------------------------------------- elementwise
__global__ void k_f32_to_f16(const float* __restrict__ in, _Float16* __restrict__ out, int n) {
  int i = blockIdx.x * blockDim.x + threadIdx.x;
  if (i < n) out[i] = (_Float16)in[i];
}

// Transposing convert: out[c*ldo + r] = (f16) in[r*cols + c].  LDS-tiled.
__global__ void k_f32_to_f16_tr(const float* __restrict__ in, _Float16* __restrict__ out,
                                int rows, int cols, int ldo) {
  __shared__ _Float16 tile[32][33];
  int c0 = blockIdx.x * 32, r0 = blockIdx.y * 32;
  int tx = threadIdx.x & 31, ty = threadIdx.x >> 5;   // 256 threads: 8 rows per pass
  for (int rr = ty; rr < 32; rr += 8) {
    int r = r0 + rr, c = c0 + tx;
    tile[rr][tx] = (r < rows && c < cols) ? (_Float16)in[(size_t)r * cols + c]
                                          : (_Float16)0.0f;
  }
  __syncthreads();
  for (int cc = ty; cc < 32; cc += 8) {
    int c = c0 + cc, r = r0 + tx;
    if (c < cols && r < rows) out[(size_t)c * ldo + r] = tile[tx][cc];
  }
}

__global__ void k_zero_f32(float* __restrict__ p, int n) {
  int i = blockIdx.x * blockDim.x + threadIdx.x;
  if (i < n) p[i] = 0.0f;
}

// ---------------------------------------------------------------- sym_norm
__global__ void k_rowsum(const float* __restrict__ A, float* __restrict__ d, int n) {
  int i = blockIdx.x;
  float s = 0.f;
  for (int j = threadIdx.x; j < n; j += TPB) {
    float a = A[(size_t)i * n + j];
    if (j == i) a += 1.0f;
    s += a;
  }
  __shared__ float red[TPB];
  red[threadIdx.x] = s;
  __syncthreads();
  for (int off = TPB / 2; off > 0; off >>= 1) {
    if (threadIdx.x < off) red[threadIdx.x] += red[threadIdx.x + off];
    __syncthreads();
  }
  if (threadIdx.x == 0) d[i] = red[0] + 1e-5f;
}

__global__ void k_symnorm(const float* __restrict__ A, const float* __restrict__ d,
                          float* __restrict__ Ahat, int n) {
  int id = blockIdx.x * blockDim.x + threadIdx.x;
  if (id >= n * n) return;
  int i = id / n, j = id % n;
  float a = A[id] + (i == j ? 1.0f : 0.0f);
  Ahat[id] = rsqrtf(d[i]) * a * rsqrtf(d[j]);
}

// ---------------------------------------------------------------- WMMA GEMM
// C[M,N](f32) = A[M,K](f16, lda) * BT[N,K](f16, ldb)^T.
// One wave computes a 32x32 tile (2x2 WMMA sub-tiles) with a software-pipelined
// K-loop: loads for k+32 are issued before the 4 WMMAs for k, so the wave only
// waits on the older load group.  Edge tiles / K-tail use the guarded path.
__global__ __launch_bounds__(32) void k_wmma_gemm(
    const _Float16* __restrict__ A, const _Float16* __restrict__ BT,
    float* __restrict__ C, int M, int N, int K,
    int lda, int ldb, int ldc, int relu)
{
  const int lane = threadIdx.x;
  const int grp  = lane >> 4;     // 0 or 1
  const int l16  = lane & 15;
  const int m0   = blockIdx.y * 32;
  const int nn0  = blockIdx.x * 32;

  v8f acc[2][2];
#pragma unroll
  for (int ti = 0; ti < 2; ++ti)
#pragma unroll
    for (int tj = 0; tj < 2; ++tj)
      acc[ti][tj] = (v8f){0.f, 0.f, 0.f, 0.f, 0.f, 0.f, 0.f, 0.f};

  const bool fullMN = (m0 + 32 <= M) && (nn0 + 32 <= N);
  int k0 = 0;

  if (fullMN) {
    // A fragment (16x32 f16): lane grp g holds K = g*8+[0..7] and g*8+16+[0..7]
    // B fragment (32x16 f16): lane grp g holds K = g*8+[0..7] and g*8+16+[0..7] of its column
    const _Float16* __restrict__ ar0 = A  + (size_t)(m0 + l16)       * lda + grp * 8;
    const _Float16* __restrict__ ar1 = A  + (size_t)(m0 + 16 + l16)  * lda + grp * 8;
    const _Float16* __restrict__ br0 = BT + (size_t)(nn0 + l16)      * ldb + grp * 8;
    const _Float16* __restrict__ br1 = BT + (size_t)(nn0 + 16 + l16) * ldb + grp * 8;
    const int Kfull = K & ~31;
    if (Kfull >= 32) {
      v8h a00 = *(const v8h*)(ar0);      v8h a01 = *(const v8h*)(ar0 + 16);
      v8h a10 = *(const v8h*)(ar1);      v8h a11 = *(const v8h*)(ar1 + 16);
      v8h b00 = *(const v8h*)(br0);      v8h b01 = *(const v8h*)(br0 + 16);
      v8h b10 = *(const v8h*)(br1);      v8h b11 = *(const v8h*)(br1 + 16);
      for (k0 = 32; k0 < Kfull; k0 += 32) {
        if (k0 + 96 < Kfull) {
          __builtin_prefetch(ar0 + k0 + 96, 0, 1);   // global_prefetch_b8
          __builtin_prefetch(br0 + k0 + 96, 0, 1);
        }
        // issue next k-step loads before consuming current fragments
        v8h na00 = *(const v8h*)(ar0 + k0);  v8h na01 = *(const v8h*)(ar0 + k0 + 16);
        v8h na10 = *(const v8h*)(ar1 + k0);  v8h na11 = *(const v8h*)(ar1 + k0 + 16);
        v8h nb00 = *(const v8h*)(br0 + k0);  v8h nb01 = *(const v8h*)(br0 + k0 + 16);
        v8h nb10 = *(const v8h*)(br1 + k0);  v8h nb11 = *(const v8h*)(br1 + k0 + 16);
        v16h af0 = __builtin_shufflevector(a00, a01, 0,1,2,3,4,5,6,7,8,9,10,11,12,13,14,15);
        v16h af1 = __builtin_shufflevector(a10, a11, 0,1,2,3,4,5,6,7,8,9,10,11,12,13,14,15);
        v16h bf0 = __builtin_shufflevector(b00, b01, 0,1,2,3,4,5,6,7,8,9,10,11,12,13,14,15);
        v16h bf1 = __builtin_shufflevector(b10, b11, 0,1,2,3,4,5,6,7,8,9,10,11,12,13,14,15);
        acc[0][0] = __builtin_amdgcn_wmma_f32_16x16x32_f16(false, af0, false, bf0,
                                                           (short)0, acc[0][0], false, false);
        acc[0][1] = __builtin_amdgcn_wmma_f32_16x16x32_f16(false, af0, false, bf1,
                                                           (short)0, acc[0][1], false, false);
        acc[1][0] = __builtin_amdgcn_wmma_f32_16x16x32_f16(false, af1, false, bf0,
                                                           (short)0, acc[1][0], false, false);
        acc[1][1] = __builtin_amdgcn_wmma_f32_16x16x32_f16(false, af1, false, bf1,
                                                           (short)0, acc[1][1], false, false);
        a00 = na00; a01 = na01; a10 = na10; a11 = na11;
        b00 = nb00; b01 = nb01; b10 = nb10; b11 = nb11;
      }
      // epilogue: consume the last fragments
      {
        v16h af0 = __builtin_shufflevector(a00, a01, 0,1,2,3,4,5,6,7,8,9,10,11,12,13,14,15);
        v16h af1 = __builtin_shufflevector(a10, a11, 0,1,2,3,4,5,6,7,8,9,10,11,12,13,14,15);
        v16h bf0 = __builtin_shufflevector(b00, b01, 0,1,2,3,4,5,6,7,8,9,10,11,12,13,14,15);
        v16h bf1 = __builtin_shufflevector(b10, b11, 0,1,2,3,4,5,6,7,8,9,10,11,12,13,14,15);
        acc[0][0] = __builtin_amdgcn_wmma_f32_16x16x32_f16(false, af0, false, bf0,
                                                           (short)0, acc[0][0], false, false);
        acc[0][1] = __builtin_amdgcn_wmma_f32_16x16x32_f16(false, af0, false, bf1,
                                                           (short)0, acc[0][1], false, false);
        acc[1][0] = __builtin_amdgcn_wmma_f32_16x16x32_f16(false, af1, false, bf0,
                                                           (short)0, acc[1][0], false, false);
        acc[1][1] = __builtin_amdgcn_wmma_f32_16x16x32_f16(false, af1, false, bf1,
                                                           (short)0, acc[1][1], false, false);
      }
      k0 = Kfull;
    }
  }

  // Guarded path: edge tiles (whole K) or K-tail of interior tiles.
  for (; k0 < K; k0 += 32) {
#pragma unroll
    for (int ti = 0; ti < 2; ++ti) {
      const int row = m0 + ti * 16 + l16;
      v16h af;
#pragma unroll
      for (int h = 0; h < 16; ++h) {
        int kk = k0 + grp * 8 + (h < 8 ? h : h + 8);
        _Float16 v = (_Float16)0.0f;
        if (row < M && kk < K) v = A[(size_t)row * lda + kk];
        af[h] = v;
      }
#pragma unroll
      for (int tj = 0; tj < 2; ++tj) {
        const int col = nn0 + tj * 16 + l16;
        v16h bf;
#pragma unroll
        for (int h = 0; h < 16; ++h) {
          int kk = k0 + grp * 8 + (h < 8 ? h : h + 8);
          _Float16 v = (_Float16)0.0f;
          if (col < N && kk < K) v = BT[(size_t)col * ldb + kk];
          bf[h] = v;
        }
        acc[ti][tj] = __builtin_amdgcn_wmma_f32_16x16x32_f16(false, af, false, bf,
                                                             (short)0, acc[ti][tj],
                                                             false, false);
      }
    }
  }

  // D layout: VGPR e -> row = grp*8 + e within sub-tile, col = l16
#pragma unroll
  for (int ti = 0; ti < 2; ++ti) {
#pragma unroll
    for (int e = 0; e < 8; ++e) {
      int r = m0 + ti * 16 + grp * 8 + e;
      if (r >= M) continue;
#pragma unroll
      for (int tj = 0; tj < 2; ++tj) {
        int col = nn0 + tj * 16 + l16;
        if (col < N) {
          float v = acc[ti][tj][e];
          if (relu) v = fmaxf(v, 0.0f);
          C[(size_t)r * ldc + col] = v;
        }
      }
    }
  }
}

// ---------------------------------------------------------------- GAT pieces
__global__ void k_gat_logits(const float* __restrict__ xW,
                             const float* __restrict__ asrc, const float* __restrict__ adst,
                             float* __restrict__ s, float* __restrict__ t,
                             int n, int H, int C) {
  int id = blockIdx.x * blockDim.x + threadIdx.x;
  if (id >= n * H) return;
  int j = id / H, h = id % H;
  const float* xr = xW + (size_t)j * H * C + (size_t)h * C;
  float ss = 0.f, tt = 0.f;
  for (int c = 0; c < C; ++c) {
    float v = xr[c];
    ss += v * asrc[h * C + c];
    tt += v * adst[h * C + c];
  }
  s[id] = ss;
  t[id] = tt;
}

// Per destination row i: online masked softmax stats (max m, sum z) over all j, all heads.
__global__ void k_gat_stats(const float* __restrict__ Ahat,
                            const float* __restrict__ s, const float* __restrict__ t,
                            float* __restrict__ m, float* __restrict__ z,
                            int n, int H) {
  int i = blockIdx.x;
  float mx[MAXH], sm[MAXH], ti[MAXH];
  for (int h = 0; h < H; ++h) { mx[h] = -1e30f; sm[h] = 0.f; ti[h] = t[i * H + h]; }
  for (int j = threadIdx.x; j < n; j += TPB) {
    float a = Ahat[(size_t)i * n + j];
    if (a > 0.f) {
      for (int h = 0; h < H; ++h) {
        float l = ti[h] + s[j * H + h];
        l = l > 0.f ? l : 0.2f * l;           // leaky relu, slope 0.2
        if (l > mx[h]) { sm[h] = sm[h] * __expf(mx[h] - l) + 1.0f; mx[h] = l; }
        else           { sm[h] += __expf(l - mx[h]); }
      }
    }
  }
  __shared__ float smx[MAXH][TPB];
  __shared__ float ssm[MAXH][TPB];
  for (int h = 0; h < H; ++h) { smx[h][threadIdx.x] = mx[h]; ssm[h][threadIdx.x] = sm[h]; }
  __syncthreads();
  for (int off = TPB / 2; off > 0; off >>= 1) {
    if (threadIdx.x < off) {
      for (int h = 0; h < H; ++h) {
        float M1 = smx[h][threadIdx.x],       Z1 = ssm[h][threadIdx.x];
        float M2 = smx[h][threadIdx.x + off], Z2 = ssm[h][threadIdx.x + off];
        float Mx = fmaxf(M1, M2);
        float Z = 0.f;
        if (M1 > -9e29f) Z += Z1 * __expf(M1 - Mx);
        if (M2 > -9e29f) Z += Z2 * __expf(M2 - Mx);
        smx[h][threadIdx.x] = Mx;
        ssm[h][threadIdx.x] = Z;
      }
    }
    __syncthreads();
  }
  if (threadIdx.x == 0)
    for (int h = 0; h < H; ++h) { m[i * H + h] = smx[h][0]; z[i * H + h] = ssm[h][0]; }
}

// Materialize attention probabilities P[h][i][j] (f16, padded row stride ldp).
__global__ void k_gat_pmat(const float* __restrict__ Ahat,
                           const float* __restrict__ s, const float* __restrict__ t,
                           const float* __restrict__ m, const float* __restrict__ z,
                           _Float16* __restrict__ P, int n, int H, int ldp) {
  long long id = (long long)blockIdx.x * blockDim.x + threadIdx.x;
  long long nn = (long long)n * n;
  if (id >= (long long)H * nn) return;
  int h = (int)(id / nn);
  long long rem = id % nn;
  int i = (int)(rem / n), j = (int)(rem % n);
  float p = 0.f;
  if (Ahat[rem] > 0.f) {
    float l = t[i * H + h] + s[j * H + h];
    l = l > 0.f ? l : 0.2f * l;
    float zz = z[i * H + h];
    p = __expf(l - m[i * H + h]) / (zz > 0.f ? zz : 1.0f);
  }
  P[((size_t)h * n + i) * ldp + j] = (_Float16)p;
}

__global__ void k_bias_relu(float* __restrict__ X, const float* __restrict__ b,
                            int total, int dim) {
  int id = blockIdx.x * blockDim.x + threadIdx.x;
  if (id >= total) return;
  X[id] = fmaxf(X[id] + b[id % dim], 0.f);
}

__global__ void k_bias_rows(float* __restrict__ X, const float* __restrict__ b,
                            int total, int dim) {
  int id = blockIdx.x * blockDim.x + threadIdx.x;
  if (id >= total) return;
  X[id] += b[id / dim];
}

// ---------------------------------------------------------------- pooling
__global__ void k_pool_scores(const float* __restrict__ X, const float* __restrict__ Wp,
                              const float* __restrict__ bp, int bi,
                              float* __restrict__ sc, int n, int dim) {
  int i = blockIdx.x * blockDim.x + threadIdx.x;
  if (i >= n) return;
  const float* xr = X + (size_t)i * dim;
  float s = 0.f;
  for (int c = 0; c < dim; ++c) s += xr[c] * Wp[c];
  s += bp[bi];
  sc[i] = 1.0f / (1.0f + __expf(-s));
}

// Exact top_k semantics (descending, stable ties by index) via rank counting.
__global__ void k_topk(const float* __restrict__ sc, int* __restrict__ idx,
                       float* __restrict__ vals, int n, int k) {
  int i = blockIdx.x;
  float si = sc[i];
  int cnt = 0;
  for (int j = threadIdx.x; j < n; j += TPB) {
    float sj = sc[j];
    if (sj > si || (sj == si && j < i)) cnt++;
  }
  __shared__ int red[TPB];
  red[threadIdx.x] = cnt;
  __syncthreads();
  for (int off = TPB / 2; off > 0; off >>= 1) {
    if (threadIdx.x < off) red[threadIdx.x] += red[threadIdx.x + off];
    __syncthreads();
  }
  if (threadIdx.x == 0) {
    int r = red[0];
    if (r < k) { idx[r] = i; vals[r] = si; }
  }
}

__global__ void k_gather_X(const float* __restrict__ X, const int* __restrict__ idx,
                           const float* __restrict__ vals, float* __restrict__ Xo,
                           int k, int dim) {
  int id = blockIdx.x * blockDim.x + threadIdx.x;
  if (id >= k * dim) return;
  int r = id / dim, c = id % dim;
  Xo[id] = X[(size_t)idx[r] * dim + c] * vals[r];
}

__global__ void k_gather_A(const float* __restrict__ A, const int* __restrict__ idx,
                           float* __restrict__ Ao, int n, int k) {
  int id = blockIdx.x * blockDim.x + threadIdx.x;
  if (id >= k * k) return;
  int r = id / k, c = id % k;
  Ao[id] = A[(size_t)idx[r] * n + idx[c]];
}

__global__ void k_scatter_rows(const float* __restrict__ X, const int* __restrict__ idx,
                               float* __restrict__ Xo, int k, int dim) {
  int id = blockIdx.x * blockDim.x + threadIdx.x;
  if (id >= k * dim) return;
  int r = id / dim, c = id % dim;
  Xo[(size_t)idx[r] * dim + c] = X[id];
}

// ---------------------------------------------------------------- host
extern "C" void kernel_launch(void* const* d_in, const int* in_sizes, int n_in,
                              void* d_out, int out_size, void* d_ws, size_t ws_size,
                              hipStream_t stream) {
  (void)in_sizes; (void)n_in; (void)out_size; (void)ws_size;
  const int DIM = 256, HH = 4, CC = 64, HB = 2, CB = 128, MUP = 4096;
  const int n0 = 2048;
  const int n1 = (int)(0.9 * n0);  // 1843
  const int n2 = (int)(0.8 * n1);  // 1474
  const int n3 = (int)(0.7 * n2);  // 1031

  const float* A_in    = (const float*)d_in[0];
  const float* X_in    = (const float*)d_in[1];
  const float* W_down  = (const float*)d_in[2];
  const float* as_down = (const float*)d_in[3];
  const float* ad_down = (const float*)d_in[4];
  const float* b_down  = (const float*)d_in[5];
  const float* W_up    = (const float*)d_in[6];
  const float* as_up   = (const float*)d_in[7];
  const float* ad_up   = (const float*)d_in[8];
  const float* b_up    = (const float*)d_in[9];
  const float* W_bot   = (const float*)d_in[10];
  const float* as_bot  = (const float*)d_in[11];
  const float* ad_bot  = (const float*)d_in[12];
  const float* b_bot   = (const float*)d_in[13];
  const float* W_pool  = (const float*)d_in[14];
  const float* b_pool  = (const float*)d_in[15];
  const float* W_ups   = (const float*)d_in[16];
  const float* b_ups   = (const float*)d_in[17];

  char* base = (char*)d_ws;
  size_t off = 0;
  auto alloc = [&](size_t bytes) -> void* {
    void* p = base + off;
    off += (bytes + 255) & ~((size_t)255);
    return p;
  };
  auto align8 = [](int v) { return (v + 7) & ~7; };

  const int nP0 = align8(n0);   // padded K strides (16B-aligned f16 rows)

  float* A0   = (float*)alloc((size_t)n0 * n0 * 4);
  float* A1   = (float*)alloc((size_t)n1 * n1 * 4);
  float* A2   = (float*)alloc((size_t)n2 * n2 * 4);
  float* A3   = (float*)alloc((size_t)n3 * n3 * 4);
  float* dsum = (float*)alloc((size_t)n0 * 4);
  float* Xa   = (float*)alloc((size_t)n0 * DIM * 4);
  float* Xb   = (float*)alloc((size_t)n0 * DIM * 4);
  float* Xc   = (float*)alloc((size_t)n0 * DIM * 4);
  _Float16* Xh  = (_Float16*)alloc((size_t)n0 * DIM * 2);
  _Float16* Wt  = (_Float16*)alloc((size_t)DIM * DIM * 2);      // W^T  (f16)
  float*    xW  = (float*)alloc((size_t)n0 * DIM * 4);
  _Float16* xWt = (_Float16*)alloc((size_t)DIM * nP0 * 2);      // xW^T (f16), ld = nP
  float* sv = (float*)alloc((size_t)n0 * MAXH * 4);
  float* tv = (float*)alloc((size_t)n0 * MAXH * 4);
  float* mv = (float*)alloc((size_t)n0 * MAXH * 4);
  float* zv = (float*)alloc((size_t)n0 * MAXH * 4);
  _Float16* P = (_Float16*)alloc((size_t)MAXH * n0 * nP0 * 2);  // ld = nP
  float* scores = (float*)alloc((size_t)n0 * 4);
  int*   idxA = (int*)alloc((size_t)n0 * 4);
  int*   idxB = (int*)alloc((size_t)n0 * 4);
  int*   idxC = (int*)alloc((size_t)n0 * 4);
  float* valA = (float*)alloc((size_t)n0 * 4);
  float* valB = (float*)alloc((size_t)n0 * 4);
  float* valC = (float*)alloc((size_t)n0 * 4);
  float*    Xup   = (float*)alloc((size_t)MUP * DIM * 4);
  _Float16* Xuph  = (_Float16*)alloc((size_t)MUP * DIM * 2);
  _Float16* XT    = (_Float16*)alloc((size_t)DIM * n0 * 2);     // X^T for upsampler
  _Float16* Wupsh = (_Float16*)alloc((size_t)MUP * n0 * 2);

  auto cdiv = [](int a, int b) { return (a + b - 1) / b; };

  // C[M,N] = A[M,K] * BT[N,K]^T    (32x32 tile per wave)
  auto gemm = [&](const _Float16* Am, const _Float16* Bt, float* Cm,
                  int M, int Nn, int K, int lda, int ldb, int ldc, int relu) {
    dim3 g((unsigned)cdiv(Nn, 32), (unsigned)cdiv(M, 32));
    k_wmma_gemm<<<g, 32, 0, stream>>>(Am, Bt, Cm, M, Nn, K, lda, ldb, ldc, relu);
  };
  auto transpose_f16 = [&](const float* in, _Float16* out, int rows, int cols, int ldo) {
    dim3 g((unsigned)cdiv(cols, 32), (unsigned)cdiv(rows, 32));
    k_f32_to_f16_tr<<<g, TPB, 0, stream>>>(in, out, rows, cols, ldo);
  };

  auto run_gat = [&](const float* Ah, const float* Xin, const float* W,
                     const float* as_, const float* ad_, const float* bb,
                     int n, int H, int C, float* Xout) {
    const int nP = align8(n);
    k_f32_to_f16<<<cdiv(n * DIM, TPB), TPB, 0, stream>>>(Xin, Xh, n * DIM);
    transpose_f16(W, Wt, DIM, DIM, DIM);                        // Wt[nn][kk] = W[kk][nn]
    gemm(Xh, Wt, xW, n, DIM, DIM, DIM, DIM, DIM, 0);            // x = X @ W
    k_gat_logits<<<cdiv(n * H, TPB), TPB, 0, stream>>>(xW, as_, ad_, sv, tv, n, H, C);
    k_gat_stats<<<n, TPB, 0, stream>>>(Ah, sv, tv, mv, zv, n, H);
    long long pc = (long long)H * n * n;
    k_gat_pmat<<<(unsigned)((pc + TPB - 1) / TPB), TPB, 0, stream>>>(Ah, sv, tv, mv, zv,
                                                                     P, n, H, nP);
    transpose_f16(xW, xWt, n, DIM, nP);                         // xWt[c][j] = x[j][c]
    for (int h = 0; h < H; ++h)                                 // out_h = P_h @ x_h
      gemm(P + (size_t)h * n * nP, xWt + (size_t)h * C * nP, Xout + h * C,
           n, C, n, nP, nP, DIM, 0);
    k_bias_relu<<<cdiv(n * DIM, TPB), TPB, 0, stream>>>(Xout, bb, n * DIM, DIM);
  };

  auto run_pool = [&](const float* Xin, const float* Ain, int li,
                      int n, int k, int* idx, float* vals, float* Xo, float* Ao) {
    k_pool_scores<<<cdiv(n, TPB), TPB, 0, stream>>>(Xin, W_pool + (size_t)li * DIM,
                                                    b_pool, li, scores, n, DIM);
    k_topk<<<n, TPB, 0, stream>>>(scores, idx, vals, n, k);
    k_gather_X<<<cdiv(k * DIM, TPB), TPB, 0, stream>>>(Xin, idx, vals, Xo, k, DIM);
    k_gather_A<<<cdiv(k * k, TPB), TPB, 0, stream>>>(Ain, idx, Ao, n, k);
  };

  // --- sym_norm
  k_rowsum<<<n0, TPB, 0, stream>>>(A_in, dsum, n0);
  k_symnorm<<<cdiv(n0 * n0, TPB), TPB, 0, stream>>>(A_in, dsum, A0, n0);

  // --- encoder
  run_gat(A0, X_in, W_down + 0 * DIM * DIM, as_down + 0 * HH * CC, ad_down + 0 * HH * CC,
          b_down + 0 * DIM, n0, HH, CC, Xa);
  run_pool(Xa, A0, 0, n0, n1, idxA, valA, Xb, A1);
  run_gat(A1, Xb, W_down + 1 * DIM * DIM, as_down + 1 * HH * CC, ad_down + 1 * HH * CC,
          b_down + 1 * DIM, n1, HH, CC, Xa);
  run_pool(Xa, A1, 1, n1, n2, idxB, valB, Xb, A2);
  run_gat(A2, Xb, W_down + 2 * DIM * DIM, as_down + 2 * HH * CC, ad_down + 2 * HH * CC,
          b_down + 2 * DIM, n2, HH, CC, Xa);
  run_pool(Xa, A2, 2, n2, n3, idxC, valC, Xb, A3);

  // --- bottleneck
  run_gat(A3, Xb, W_bot, as_bot, ad_bot, b_bot, n3, HB, CB, Xa);

  // --- decoder (scatter-unpool then GAT)
  k_zero_f32<<<cdiv(n2 * DIM, TPB), TPB, 0, stream>>>(Xc, n2 * DIM);
  k_scatter_rows<<<cdiv(n3 * DIM, TPB), TPB, 0, stream>>>(Xa, idxC, Xc, n3, DIM);
  run_gat(A2, Xc, W_up + 0 * DIM * DIM, as_up + 0 * HH * CC, ad_up + 0 * HH * CC,
          b_up + 0 * DIM, n2, HH, CC, Xb);

  k_zero_f32<<<cdiv(n1 * DIM, TPB), TPB, 0, stream>>>(Xc, n1 * DIM);
  k_scatter_rows<<<cdiv(n2 * DIM, TPB), TPB, 0, stream>>>(Xb, idxB, Xc, n2, DIM);
  run_gat(A1, Xc, W_up + 1 * DIM * DIM, as_up + 1 * HH * CC, ad_up + 1 * HH * CC,
          b_up + 1 * DIM, n1, HH, CC, Xa);

  k_zero_f32<<<cdiv(n0 * DIM, TPB), TPB, 0, stream>>>(Xc, n0 * DIM);
  k_scatter_rows<<<cdiv(n1 * DIM, TPB), TPB, 0, stream>>>(Xa, idxA, Xc, n1, DIM);
  run_gat(A0, Xc, W_up + 2 * DIM * DIM, as_up + 2 * HH * CC, ad_up + 2 * HH * CC,
          b_up + 2 * DIM, n0, HH, CC, Xb);

  // --- upsampler: X_up = W_ups @ X + b_ups[:,None]; A_up = relu(X_up @ X_up^T)
  k_f32_to_f16<<<cdiv(MUP * n0, TPB), TPB, 0, stream>>>(W_ups, Wupsh, MUP * n0);
  transpose_f16(Xb, XT, n0, DIM, n0);                           // XT[c][j] = X[j][c]
  gemm(Wupsh, XT, Xup, MUP, DIM, n0, n0, n0, DIM, 0);
  k_bias_rows<<<cdiv(MUP * DIM, TPB), TPB, 0, stream>>>(Xup, b_ups, MUP * DIM, DIM);
  k_f32_to_f16<<<cdiv(MUP * DIM, TPB), TPB, 0, stream>>>(Xup, Xuph, MUP * DIM);
  gemm(Xuph, Xuph, (float*)d_out, MUP, MUP, DIM, DIM, DIM, MUP, /*relu=*/1);
}
